// MyGRU_84920093376825
// MI455X (gfx1250) — matrix operand order repaired and checked by reading, not compiled, not run on hardware
//
#include <hip/hip_runtime.h>
#include <hip/hip_bf16.h>
#include <math.h>

#define D 300
#define T 2048
#define V 50000
#define NB 20      // persistent blocks for the scan
#define RPB 15     // rows of each recurrent matrix owned per block (NB*RPB == D)

typedef float v2f __attribute__((ext_vector_type(2)));
typedef float v8f __attribute__((ext_vector_type(8)));
typedef __bf16 v8bf  __attribute__((ext_vector_type(8)));
typedef __bf16 v16bf __attribute__((ext_vector_type(16)));
typedef int   v4i  __attribute__((ext_vector_type(4)));

#if defined(__has_builtin)
#  if __has_builtin(__builtin_amdgcn_global_load_async_to_lds_b128)
#    define HAVE_ASYNC_LDS 1
#  endif
#endif

#ifdef HAVE_ASYNC_LDS
typedef __attribute__((address_space(1))) v4i g_v4i;   // global int4
typedef __attribute__((address_space(3))) v4i l_v4i;   // LDS int4
// per-lane: copy 16B global -> LDS via the async DMA path (tracked by ASYNCcnt)
__device__ __forceinline__ void async_vec16(float* dst, const float* src, int j) {
  __builtin_amdgcn_global_load_async_to_lds_b128(
      (g_v4i*)(src + 4 * j), (l_v4i*)(dst + 4 * j), 0, 0);
}
__device__ __forceinline__ void async_wait0() {
#  if __has_builtin(__builtin_amdgcn_s_wait_asynccnt)
  __builtin_amdgcn_s_wait_asynccnt(0);
#  else
  asm volatile("s_wait_asynccnt 0x0" ::: "memory");
#  endif
}
#endif

// ---------------------------------------------------------------------------
// init: zero h-state vectors, grid-barrier counters, and predictions_senses
// ---------------------------------------------------------------------------
__global__ void k_init(float* gH1, float* gH2, unsigned* bar, float* pred) {
  int i = blockIdx.x * 256 + threadIdx.x;
  if (i < D) { gH1[i] = 0.f; gH2[i] = 0.f; }
  if (i < 2) bar[i] = 0u;
  if (i < T) pred[i] = 0.f;
}

// ---------------------------------------------------------------------------
// gather: xs[t, :] = X[idx[t], :]
// ---------------------------------------------------------------------------
__global__ void k_gather(const int* __restrict__ idx, const float* __restrict__ X,
                         float* __restrict__ xs) {
  int i = blockIdx.x * 256 + threadIdx.x;
  if (i < T * D) {
    int t = i / D, d = i - t * D;
    xs[i] = X[(size_t)idx[t] * D + d];
  }
}

// ---------------------------------------------------------------------------
// Exact fp32 WMMA GEMM (V_WMMA_F32_16X16X4_F32) for the small precompute
// GEMMs: C[M,N] = A[M,K] * W[N,K]^T (+ bias). Block tile 64x64, 8 waves 4x2.
// ---------------------------------------------------------------------------
__global__ __launch_bounds__(256)
void k_gemm_f32(const float* __restrict__ A, const float* __restrict__ W,
                const float* __restrict__ bias, float* __restrict__ C,
                int M, int N, int K, int ldc) {
  __shared__ float sA[64][34];
  __shared__ float sB[64][34];
  const int tid  = threadIdx.x;
  const int lane = tid & 31;
  const int wave = tid >> 5;
  const int m0 = blockIdx.y * 64, n0 = blockIdx.x * 64;
  const int m_off = (wave >> 1) * 16;
  const int n_off = (wave & 1) * 32;
  const int hi = lane >> 4, lo = lane & 15;

  v8f acc0 = {0.f,0.f,0.f,0.f,0.f,0.f,0.f,0.f};
  v8f acc1 = {0.f,0.f,0.f,0.f,0.f,0.f,0.f,0.f};

  for (int kc = 0; kc < K; kc += 32) {
    for (int x = tid; x < 64 * 32; x += 256) {
      int r = x >> 5, c = x & 31;
      int gk = kc + c;
      sA[r][c] = (gk < K) ? A[(size_t)(m0 + r) * K + gk] : 0.f;
      int gn = n0 + r;
      sB[r][c] = (gn < N && gk < K) ? W[(size_t)gn * K + gk] : 0.f;
    }
    __syncthreads();
#pragma unroll
    for (int kk = 0; kk < 32; kk += 4) {
      v2f a, b0, b1;
      a.x  = sA[m_off + lo][kk + 2 * hi];
      a.y  = sA[m_off + lo][kk + 2 * hi + 1];
      b0.x = sB[n_off + lo][kk + 2 * hi];
      b0.y = sB[n_off + lo][kk + 2 * hi + 1];
      b1.x = sB[n_off + 16 + lo][kk + 2 * hi];
      b1.y = sB[n_off + 16 + lo][kk + 2 * hi + 1];
      acc0 = __builtin_amdgcn_wmma_f32_16x16x4_f32(false, a, false, b0, (short)0, acc0, false, false);
      acc1 = __builtin_amdgcn_wmma_f32_16x16x4_f32(false, a, false, b1, (short)0, acc1, false, false);
    }
    __syncthreads();
  }

  const int nb = n0 + n_off + lo;
  float bv0 = (bias && nb < N) ? bias[nb] : 0.f;
  float bv1 = (bias && nb + 16 < N) ? bias[nb + 16] : 0.f;
#pragma unroll
  for (int i = 0; i < 8; ++i) {
    int m = m0 + m_off + i + 8 * hi;
    if (nb < N)      C[(size_t)m * ldc + nb]      = acc0[i] + bv0;
    if (nb + 16 < N) C[(size_t)m * ldc + nb + 16] = acc1[i] + bv1;
  }
}

// ---------------------------------------------------------------------------
// bf16 WMMA GEMM (V_WMMA_F32_16X16X32_BF16) for the huge vocab projection.
// fp32 operands are converted to bf16 on the fly while staging into LDS
// (K tail 300->320 zero-padded in LDS). Block tile 128x64: 8 waves along M,
// each wave computes a 16x64 strip = 4 accumulators. fp32 accumulate.
// LDS row stride 56 elems (112B): 16B-aligned b128 frag loads, bank-spread.
// Logits are written with non-temporal stores (410MB stream >> L2).
// ---------------------------------------------------------------------------
__global__ __launch_bounds__(256)
void k_gemm_bf16(const float* __restrict__ A, const float* __restrict__ W,
                 const float* __restrict__ bias, float* __restrict__ C,
                 int M, int N, int K, int ldc) {
  __shared__ __bf16 sA[128][56];
  __shared__ __bf16 sB[64][56];
  const int tid  = threadIdx.x;
  const int lane = tid & 31;
  const int wave = tid >> 5;
  const int hi = lane >> 4, lo = lane & 15;
  const int m0 = blockIdx.y * 128, n0 = blockIdx.x * 64;
  const int m_off = wave * 16;

  v8f zero = {0.f,0.f,0.f,0.f,0.f,0.f,0.f,0.f};
  v8f acc[4] = {zero, zero, zero, zero};

  for (int kc = 0; kc < K; kc += 32) {
    for (int x = tid; x < 128 * 32; x += 256) {
      int r = x >> 5, c = x & 31;
      int gk = kc + c;
      float av = (gk < K) ? A[(size_t)(m0 + r) * K + gk] : 0.f;
      sA[r][c] = (__bf16)av;
    }
    for (int x = tid; x < 64 * 32; x += 256) {
      int r = x >> 5, c = x & 31;
      int gk = kc + c, gn = n0 + r;
      float wv = (gn < N && gk < K) ? W[(size_t)gn * K + gk] : 0.f;
      sB[r][c] = (__bf16)wv;
    }
    __syncthreads();

    // A frag: lane(lo,hi) -> rows m_off+lo, K = {8*hi..8*hi+7, 16+8*hi..+7}
    v8bf alo = *(const v8bf*)(&sA[m_off + lo][8 * hi]);
    v8bf ahi = *(const v8bf*)(&sA[m_off + lo][16 + 8 * hi]);
    v16bf a = __builtin_shufflevector(alo, ahi, 0,1,2,3,4,5,6,7,8,9,10,11,12,13,14,15);
#pragma unroll
    for (int f = 0; f < 4; ++f) {
      // B frag: lane(lo,hi) -> W-row 16f+lo, K = {8*hi..8*hi+7, 16+8*hi..+7}
      v8bf blo = *(const v8bf*)(&sB[16 * f + lo][8 * hi]);
      v8bf bhi = *(const v8bf*)(&sB[16 * f + lo][16 + 8 * hi]);
      v16bf b = __builtin_shufflevector(blo, bhi, 0,1,2,3,4,5,6,7,8,9,10,11,12,13,14,15);
      acc[f] = __builtin_amdgcn_wmma_f32_16x16x32_bf16(false, a, false, b, (short)0, acc[f], false, false);
    }
    __syncthreads();
  }

#pragma unroll
  for (int f = 0; f < 4; ++f) {
    int n = n0 + 16 * f + lo;
    if (n < N) {
      float bv = bias ? bias[n] : 0.f;
#pragma unroll
      for (int i = 0; i < 8; ++i) {
        int m = m0 + m_off + i + 8 * hi;
        __builtin_nontemporal_store(acc[f][i] + bv, &C[(size_t)m * ldc + n]);
      }
    }
  }
}

// ---------------------------------------------------------------------------
// Sequential 2-layer GRU scan: persistent NB blocks, 4 device barriers/step.
// Each block owns RPB rows of every recurrent matrix (weights stay in L2).
// Cross-step broadcast vectors are pulled into LDS via async load-to-LDS
// when the toolchain exposes it.
// ---------------------------------------------------------------------------
__device__ __forceinline__ float wdot(const float* __restrict__ row,
                                      const float* __restrict__ sv) {
  int lane = threadIdx.x & 31;
  float s = 0.f;
  for (int k = lane; k < D; k += 32) s = fmaf(row[k], sv[k], s);
#pragma unroll
  for (int off = 16; off > 0; off >>= 1) s += __shfl_xor(s, off);
  return s;
}

__device__ __forceinline__ void grid_barrier(unsigned* bar) {
  __threadfence();
  __syncthreads();
  if (threadIdx.x == 0) {
    volatile unsigned* gen = bar + 1;
    unsigned g = *gen;
    if (atomicAdd(bar, 1u) == NB - 1) {
      bar[0] = 0u;
      __threadfence();
      atomicAdd((unsigned*)gen, 1u);
    } else {
      while (*gen == g) __builtin_amdgcn_s_sleep(2);
    }
  }
  __syncthreads();
}

__device__ __forceinline__ float sigmoidf_(float x) {
  return 1.f / (1.f + __expf(-x));
}

__global__ __launch_bounds__(256)
void k_gru_scan(const float* __restrict__ PZ, const float* __restrict__ PR,
                const float* __restrict__ PW,
                const float* __restrict__ Uz1, const float* __restrict__ Ur1,
                const float* __restrict__ U1,  const float* __restrict__ bU1,
                const float* __restrict__ Wz2, const float* __restrict__ Uz2,
                const float* __restrict__ Wr2, const float* __restrict__ Ur2,
                const float* __restrict__ W2,  const float* __restrict__ bW2,
                const float* __restrict__ U2,  const float* __restrict__ bU2,
                float* gH1, float* gH2, float* gRH1, float* gRH2,
                float* __restrict__ h2seq, unsigned* bar) {
  __shared__ float sH1[D], sH2[D], sX[D], sZ[RPB];
  const int tid = threadIdx.x, wave = tid >> 5, lane = tid & 31;
  const int d0 = blockIdx.x * RPB;
  volatile float* vH1 = gH1;
  volatile float* vH2 = gH2;
  volatile float* vR1 = gRH1;
  volatile float* vR2 = gRH2;
  (void)vH1; (void)vH2; (void)vR1; (void)vR2;

  for (int t = 0; t < T; ++t) {
    const float* pz = PZ + t * D;
    const float* pr = PR + t * D;
    const float* pw = PW + t * D;

    // load h1, h2 from previous step
#ifdef HAVE_ASYNC_LDS
    if (tid < D / 4) { async_vec16(sH1, gH1, tid); async_vec16(sH2, gH2, tid); }
    async_wait0();
#else
    for (int j = tid; j < D; j += 256) { sH1[j] = vH1[j]; sH2[j] = vH2[j]; }
#endif
    __syncthreads();

    // phase 1: z1, r1 rows; publish r1*h1
    for (int i = wave; i < RPB; i += 8) {
      int d = d0 + i;
      float az = pz[d] + wdot(Uz1 + (size_t)d * D, sH1);
      float ar = pr[d] + wdot(Ur1 + (size_t)d * D, sH1);
      if (lane == 0) {
        sZ[i] = sigmoidf_(az);
        gRH1[d] = sigmoidf_(ar) * sH1[d];
      }
    }
    grid_barrier(bar);
#ifdef HAVE_ASYNC_LDS
    if (tid < D / 4) async_vec16(sX, gRH1, tid);
    async_wait0();
#else
    for (int j = tid; j < D; j += 256) sX[j] = vR1[j];   // full r1*h1
#endif
    __syncthreads();

    // phase 2: ht1, h1n rows (PW already contains x@W1^T + bW1)
    for (int i = wave; i < RPB; i += 8) {
      int d = d0 + i;
      float ht = tanhf(pw[d] + bU1[d] + wdot(U1 + (size_t)d * D, sX));
      if (lane == 0) {
        float z = sZ[i];
        gH1[d] = z * ht + (1.f - z) * sH1[d];
      }
    }
    grid_barrier(bar);
#ifdef HAVE_ASYNC_LDS
    if (tid < D / 4) async_vec16(sX, gH1, tid);
    async_wait0();
#else
    for (int j = tid; j < D; j += 256) sX[j] = vH1[j];   // full h1n
#endif
    __syncthreads();

    // phase 3: z2, r2 rows; publish r2*h2
    for (int i = wave; i < RPB; i += 8) {
      int d = d0 + i;
      float az = wdot(Wz2 + (size_t)d * D, sX) + wdot(Uz2 + (size_t)d * D, sH2);
      float ar = wdot(Wr2 + (size_t)d * D, sX) + wdot(Ur2 + (size_t)d * D, sH2);
      if (lane == 0) {
        sZ[i] = sigmoidf_(az);
        gRH2[d] = sigmoidf_(ar) * sH2[d];
      }
    }
    grid_barrier(bar);
#ifdef HAVE_ASYNC_LDS
    if (tid < D / 4) async_vec16(sH1, gRH2, tid);        // reuse sH1 for r2*h2
    async_wait0();
#else
    for (int j = tid; j < D; j += 256) sH1[j] = vR2[j];
#endif
    __syncthreads();

    // phase 4: ht2, h2n rows; emit h2seq[t]
    for (int i = wave; i < RPB; i += 8) {
      int d = d0 + i;
      float ht = tanhf(wdot(W2 + (size_t)d * D, sX) + bW2[d] +
                       wdot(U2 + (size_t)d * D, sH1) + bU2[d]);
      if (lane == 0) {
        float z = sZ[i];
        float h2n = z * ht + (1.f - z) * sH2[d];
        gH2[d] = h2n;
        h2seq[(size_t)t * D + d] = h2n;
      }
    }
    grid_barrier(bar);
  }
}

// ---------------------------------------------------------------------------
// Row-wise online logsumexp over V=50000: lse[t] = max + log(sum exp(x-max))
// ---------------------------------------------------------------------------
__global__ __launch_bounds__(256)
void k_rowstat(const float* __restrict__ logits, float* __restrict__ lse) {
  int t = blockIdx.x;
  const float* row = logits + (size_t)t * V;
  float m = -3.0e38f, s = 0.f;
  for (int v = threadIdx.x; v < V; v += 256) {
    float x = __builtin_nontemporal_load(&row[v]);
    float nm = fmaxf(m, x);
    s = s * __expf(m - nm) + __expf(x - nm);
    m = nm;
  }
#pragma unroll
  for (int off = 16; off > 0; off >>= 1) {
    float mo = __shfl_xor(m, off), so = __shfl_xor(s, off);
    float nm = fmaxf(m, mo);
    s = s * __expf(m - nm) + so * __expf(mo - nm);
    m = nm;
  }
  __shared__ float sm[8], ss[8];
  int wave = threadIdx.x >> 5, lane = threadIdx.x & 31;
  if (lane == 0) { sm[wave] = m; ss[wave] = s; }
  __syncthreads();
  if (threadIdx.x == 0) {
    m = sm[0]; s = ss[0];
    for (int w = 1; w < 8; ++w) {
      float nm = fmaxf(m, sm[w]);
      s = s * __expf(m - nm) + ss[w] * __expf(sm[w] - nm);
      m = nm;
    }
    lse[t] = m + __logf(s);
  }
}

// logp = logits - lse[t], in place on d_out (non-temporal stream)
__global__ void k_logsub(float* __restrict__ out, const float* __restrict__ lse) {
  int t = blockIdx.y;
  int v = blockIdx.x * 256 + threadIdx.x;
  if (v < V) {
    float* p = &out[(size_t)t * V + v];
    __builtin_nontemporal_store(__builtin_nontemporal_load(p) - lse[t], p);
  }
}

// ---------------------------------------------------------------------------
extern "C" void kernel_launch(void* const* d_in, const int* in_sizes, int n_in,
                              void* d_out, int out_size, void* d_ws, size_t ws_size,
                              hipStream_t stream) {
  (void)in_sizes; (void)n_in; (void)out_size; (void)ws_size;
  const int*   idx = (const int*)  d_in[0];
  const float* X   = (const float*)d_in[1];
  const float* Wz1 = (const float*)d_in[2];
  const float* Uz1 = (const float*)d_in[3];
  const float* Wr1 = (const float*)d_in[4];
  const float* Ur1 = (const float*)d_in[5];
  const float* W1  = (const float*)d_in[6];
  const float* bW1 = (const float*)d_in[7];
  const float* U1  = (const float*)d_in[8];
  const float* bU1 = (const float*)d_in[9];
  const float* Wz2 = (const float*)d_in[10];
  const float* Uz2 = (const float*)d_in[11];
  const float* Wr2 = (const float*)d_in[12];
  const float* Ur2 = (const float*)d_in[13];
  const float* W2  = (const float*)d_in[14];
  const float* bW2 = (const float*)d_in[15];
  const float* U2  = (const float*)d_in[16];
  const float* bU2 = (const float*)d_in[17];
  const float* Wg  = (const float*)d_in[18];
  const float* bg  = (const float*)d_in[19];

  float* out = (float*)d_out;
  float* ws  = (float*)d_ws;
  float* xs   = ws;              // T*D
  float* PZ   = xs  + (size_t)T * D;
  float* PR   = PZ  + (size_t)T * D;
  float* PW   = PR  + (size_t)T * D;
  float* h2s  = PW  + (size_t)T * D;
  float* gH1  = h2s + (size_t)T * D;
  float* gH2  = gH1 + D;
  float* gRH1 = gH2 + D;
  float* gRH2 = gRH1 + D;
  float* lse  = gRH2 + D;        // T
  unsigned* bar = (unsigned*)(lse + T);

  k_init<<<8, 256, 0, stream>>>(gH1, gH2, bar, out + (size_t)T * V);
  k_gather<<<(T * D + 255) / 256, 256, 0, stream>>>(idx, X, xs);

  dim3 g1((D + 63) / 64, T / 64);
  k_gemm_f32<<<g1, 256, 0, stream>>>(xs, Wz1, nullptr, PZ, T, D, D, D);
  k_gemm_f32<<<g1, 256, 0, stream>>>(xs, Wr1, nullptr, PR, T, D, D, D);
  k_gemm_f32<<<g1, 256, 0, stream>>>(xs, W1,  bW1,     PW, T, D, D, D);

  k_gru_scan<<<NB, 256, 0, stream>>>(PZ, PR, PW, Uz1, Ur1, U1, bU1,
                                     Wz2, Uz2, Wr2, Ur2, W2, bW2, U2, bU2,
                                     gH1, gH2, gRH1, gRH2, h2s, bar);

  dim3 g2((V + 63) / 64, T / 128);
  k_gemm_bf16<<<g2, 256, 0, stream>>>(h2s, Wg, bg, out, T, V, D, V);

  k_rowstat<<<T, 256, 0, stream>>>(out, lse);
  dim3 g3((V + 255) / 256, T);
  k_logsub<<<g3, 256, 0, stream>>>(out, lse);
}
